// GPTQLinear_31095563223123
// MI455X (gfx1250) — matrix operand order repaired and checked by reading, not compiled, not run on hardware
//
#include <hip/hip_runtime.h>

typedef __attribute__((ext_vector_type(16))) _Float16 v16h;
typedef __attribute__((ext_vector_type(8)))  _Float16 h8;
typedef __attribute__((ext_vector_type(8)))  float    v8f;
typedef __attribute__((ext_vector_type(4)))  float    f32x4;
typedef __attribute__((ext_vector_type(4)))  int      i32x4;

#define K_DIM   4096
#define N_DIM   11008
#define GROUP   128
#define NGROUPS (K_DIM / GROUP)          /* 32 */
#define MT      8                        /* 16-row M tiles per wave      */
#define NT      2                        /* 16-col N tiles per wave      */
#define WAVES   4                        /* waves per block              */
#define BLOCK_M (MT * 16)                /* 128 rows per block           */
#define BLOCK_N (NT * 16 * WAVES)        /* 128 cols per block           */

/* ---- pre-pass: x f32 -> f16 into workspace (8 elems / thread) ---- */
__global__ __launch_bounds__(256) void
cvt_x_to_f16(const float* __restrict__ x, _Float16* __restrict__ xh, int n)
{
    const int i = (blockIdx.x * blockDim.x + threadIdx.x) * 8;
    if (i + 8 <= n) {
        f32x4 a = *(const f32x4*)(x + i);
        f32x4 b = *(const f32x4*)(x + i + 4);
        h8 h;
#pragma unroll
        for (int j = 0; j < 4; ++j) {
            h[j]     = (_Float16)a[j];
            h[4 + j] = (_Float16)b[j];
        }
        *(h8*)(xh + i) = h;
    }
}

/* ---- main GEMM: AF16 selects f16 A from workspace vs inline f32 cvt ---- */
template<bool AF16>
__global__ __launch_bounds__(WAVES * 32) void
gptq_wmma_f16(const _Float16* __restrict__ xh,
              const float*    __restrict__ xf,
              const int*      __restrict__ qw,
              const float*    __restrict__ scales,
              const float*    __restrict__ bias,
              float*          __restrict__ out)
{
    const int lane = threadIdx.x & 31;
    const int wave = threadIdx.x >> 5;
    const int l15  = lane & 15;
    const int lhi  = lane >> 4;          /* 0: lanes 0-15, 1: lanes 16-31 */

    const int mbase = blockIdx.y * BLOCK_M;
    const int nbase = blockIdx.x * BLOCK_N + wave * (NT * 16);

    /* A addressing (16-bit 16x32 A layout: lanes 16-31 offset by 8 in K;
       each lane covers K ranges [0,8) and [16,24) of the 32-step). */
    const _Float16* xrh[MT];
    const float*    xrf[MT];
#pragma unroll
    for (int mt = 0; mt < MT; ++mt) {
        const size_t row = (size_t)(mbase + mt * 16 + l15) * K_DIM + lhi * 8;
        xrh[mt] = xh + row;
        xrf[mt] = xf + row;
    }

    /* B addressing: lane l15 owns weight row o = B column n; lanes 16-31
       offset by 8 in K; each lane covers K [0,8) and [16,24) of the step. */
    const int* qrow[NT];
    int        ocol[NT];
#pragma unroll
    for (int nt = 0; nt < NT; ++nt) {
        ocol[nt] = nbase + nt * 16 + l15;
        qrow[nt] = qw + (size_t)ocol[nt] * K_DIM + lhi * 8;
    }

    v8f tot[MT][NT] = {};                /* running f32 output tiles */

    for (int g = 0; g < NGROUPS; ++g) {
        v8f acc[MT][NT] = {};            /* per-group partial sums   */

#pragma unroll
        for (int kk = 0; kk < 4; ++kk) { /* 4 x K=32 per 128-group   */
            const int k = g * GROUP + kk * 32;

            /* ---- MT A fragments ---- */
            v16h A[MT];
#pragma unroll
            for (int mt = 0; mt < MT; ++mt) {
                if constexpr (AF16) {
                    h8 lo = *(const h8*)(xrh[mt] + k);        /* K [0,8)   */
                    h8 hi = *(const h8*)(xrh[mt] + k + 16);   /* K [16,24) */
                    A[mt] = __builtin_shufflevector(lo, hi,
                              0, 1, 2, 3, 4, 5, 6, 7,
                              8, 9, 10, 11, 12, 13, 14, 15);
                } else {
                    f32x4 a0 = *(const f32x4*)(xrf[mt] + k);
                    f32x4 a1 = *(const f32x4*)(xrf[mt] + k + 4);
                    f32x4 a2 = *(const f32x4*)(xrf[mt] + k + 16);
                    f32x4 a3 = *(const f32x4*)(xrf[mt] + k + 20);
#pragma unroll
                    for (int i = 0; i < 4; ++i) {
                        A[mt][i]      = (_Float16)a0[i];
                        A[mt][4 + i]  = (_Float16)a1[i];
                        A[mt][8 + i]  = (_Float16)a2[i];
                        A[mt][12 + i] = (_Float16)a3[i];
                    }
                }
            }

            /* ---- NT B fragments: int loads -> v_cvt_f16_i16 ---- */
            v16h Bf[NT];
#pragma unroll
            for (int nt = 0; nt < NT; ++nt) {
                const int* qp = qrow[nt] + k;
                i32x4 q0 = *(const i32x4*)(qp);
                i32x4 q1 = *(const i32x4*)(qp + 4);
                i32x4 q2 = *(const i32x4*)(qp + 16);
                i32x4 q3 = *(const i32x4*)(qp + 20);
#pragma unroll
                for (int i = 0; i < 4; ++i) {
                    Bf[nt][i]      = (_Float16)(short)q0[i]; /* exact: 0..15 */
                    Bf[nt][4 + i]  = (_Float16)(short)q1[i];
                    Bf[nt][8 + i]  = (_Float16)(short)q2[i];
                    Bf[nt][12 + i] = (_Float16)(short)q3[i];
                }
            }

            /* ---- 16 WMMAs: every A x every B ---- */
#pragma unroll
            for (int mt = 0; mt < MT; ++mt)
#pragma unroll
                for (int nt = 0; nt < NT; ++nt)
                    acc[mt][nt] = __builtin_amdgcn_wmma_f32_16x16x32_f16(
                        /*neg_a=*/false, A[mt], /*neg_b=*/false, Bf[nt],
                        /*c_mod=*/(short)0, acc[mt][nt],
                        /*reuse_a=*/false, /*reuse_b=*/false);
        }

        /* ---- fold group dequant scale (per-lane scalar, per N column) ---- */
#pragma unroll
        for (int nt = 0; nt < NT; ++nt) {
            const float sc = scales[(size_t)ocol[nt] * NGROUPS + g];
#pragma unroll
            for (int mt = 0; mt < MT; ++mt)
                tot[mt][nt] += acc[mt][nt] * sc;
        }

        /* prefetch next group's weight lines */
        if (g + 1 < NGROUPS) {
#pragma unroll
            for (int nt = 0; nt < NT; ++nt)
                __builtin_prefetch(qrow[nt] + (g + 1) * GROUP, 0, 1);
        }
    }

    /* ---- epilogue: bias + store (D layout: m = r + 8*lhi, n = l15) ---- */
#pragma unroll
    for (int nt = 0; nt < NT; ++nt) {
        const float bv = bias[ocol[nt]];
#pragma unroll
        for (int mt = 0; mt < MT; ++mt) {
            float* op = out + (size_t)(mbase + mt * 16 + lhi * 8) * N_DIM
                            + ocol[nt];
#pragma unroll
            for (int r = 0; r < 8; ++r)
                op[(size_t)r * N_DIM] = tot[mt][nt][r] + bv;
        }
    }
}

extern "C" void kernel_launch(void* const* d_in, const int* in_sizes, int n_in,
                              void* d_out, int out_size, void* d_ws, size_t ws_size,
                              hipStream_t stream) {
    (void)n_in; (void)out_size;

    const float* x      = (const float*)d_in[0];   /* (B,S,IN)  f32   */
    const int*   qw     = (const int*)d_in[1];     /* (OUT,IN)  int32 */
    const float* scales = (const float*)d_in[2];   /* (OUT,32)  f32   */
    const float* bias   = (const float*)d_in[3];   /* (OUT,)    f32   */
    float*       out    = (float*)d_out;           /* (B,S,OUT) f32   */

    const int nx = in_sizes[0];                    /* B*S*IN = 2 Mi   */
    const int M  = nx / K_DIM;                     /* B*S = 512       */
    dim3 grid(N_DIM / BLOCK_N, M / BLOCK_M);       /* (86, 4)         */

    if (ws_size >= (size_t)nx * sizeof(_Float16)) {
        _Float16* xh = (_Float16*)d_ws;
        cvt_x_to_f16<<<(nx / 8 + 255) / 256, 256, 0, stream>>>(x, xh, nx);
        gptq_wmma_f16<true><<<grid, WAVES * 32, 0, stream>>>(
            xh, x, qw, scales, bias, out);
    } else {
        gptq_wmma_f16<false><<<grid, WAVES * 32, 0, stream>>>(
            nullptr, x, qw, scales, bias, out);
    }
}
